// TP_2D_89610197664464
// MI455X (gfx1250) — matrix operand order repaired and checked by reading, not compiled, not run on hardware
//
#include <hip/hip_runtime.h>
#include <hip/hip_bf16.h>

// ---------------- types ----------------
typedef __attribute__((ext_vector_type(8)))  float    v8f;
typedef __attribute__((ext_vector_type(8)))  _Float16 v8h;
typedef __attribute__((ext_vector_type(16))) _Float16 v16h;

#define BB   256
#define CC   128
#define TT   64
#define HH1  1024
#define HH2  1024
#define NCC  10
#define BTILE 16   // batch rows per workgroup
#define NWAVE 16   // waves per workgroup; each wave owns 64 H columns

// LDS layout (dynamic, byte offsets)
#define LDS_S1    0                 // [16][1024] f16 spikes layer 1 (32 KB)
#define LDS_S2    (32 * 1024)       // [16][1024] f16 spikes layer 2 (32 KB)
#define LDS_XBUF  (64 * 1024)       // double-buffered X tile, 2 x 4 KB f16
#define LDS_L3    (72 * 1024)       // 256 f32 layer-3 partial accumulator (1 KB)
#define LDS_BYTES (73 * 1024)

// ------------- precompute kernels -------------
// xT16[t][b][c] = (f16) x[b][c][t]  -- contiguous 4KB tile per (t, 16-row batch tile)
__global__ void snn_transpose_x(const float* __restrict__ x, _Float16* __restrict__ xT16) {
    int idx = blockIdx.x * 256 + threadIdx.x;           // over B*C*T, t fastest in src
    if (idx >= BB * CC * TT) return;
    int t = idx % TT;
    int c = (idx / TT) % CC;
    int b = idx / (TT * CC);
    xT16[((size_t)t * BB + b) * CC + c] = (_Float16)x[idx];
}

__global__ void snn_cvt16(const float* __restrict__ w, _Float16* __restrict__ w16, int n) {
    int i = blockIdx.x * 256 + threadIdx.x;
    if (i < n) w16[i] = (_Float16)w[i];
}

// out_w padded from [10,1024] to [16,1024], zero rows 10..15
__global__ void snn_build_ow(const float* __restrict__ ow, _Float16* __restrict__ ow16) {
    int i = blockIdx.x * 256 + threadIdx.x;             // over 16*1024
    if (i >= 16 * HH2) return;
    int n = i >> 10, k = i & 1023;
    ow16[i] = (n < NCC) ? (_Float16)ow[n * HH2 + k] : (_Float16)0.0f;
}

// ------------- fused SNN kernel -------------
// grid = 16 WGs (one per 16-row batch tile), block = 512 threads = 16 wave32 waves.
// Membrane state lives in WMMA accumulator registers for all 64 steps; spikes cross
// layers via LDS; X tiles arrive via async global->LDS copies double-buffered one
// step ahead (issued by wave 15); layer 3 is split over waves 0..3 with ds_add_f32
// partials consumed by wave 0 one step late (no extra barrier).
__global__ void __launch_bounds__(512)
snn_fused(const _Float16* __restrict__ xT16,
          const _Float16* __restrict__ enc16, const float* __restrict__ enc_b,
          const _Float16* __restrict__ hid16, const float* __restrict__ hid_b,
          const _Float16* __restrict__ ow16,  const float* __restrict__ out_b,
          float* __restrict__ out)
{
    extern __shared__ char lds_raw[];
    _Float16* s1    = (_Float16*)(lds_raw + LDS_S1);
    _Float16* s2    = (_Float16*)(lds_raw + LDS_S2);
    float*    l3acc = (float*)(lds_raw + LDS_L3);

    const int tid   = threadIdx.x;
    const int wave  = tid >> 5;
    const int lane  = tid & 31;
    const int nl    = lane & 15;        // N column in tile; also M row for A frags
    const int half  = lane >> 4;        // lane-half select
    const int mbase = half * 8;         // C/D rows covered by this lane-half
    const int h0    = wave * 64;        // this wave's H-column base
    const int b0    = blockIdx.x * BTILE;

    float bias1[4], bias2[4];
#pragma unroll
    for (int nt = 0; nt < 4; ++nt) {
        bias1[nt] = enc_b[h0 + nt * 16 + nl];
        bias2[nt] = hid_b[h0 + nt * 16 + nl];
    }
    const float bias3 = (wave == 0 && nl < NCC) ? out_b[nl] : 0.0f;

    v8f v1s[4] = {}, v2s[4] = {};
    v8f v3 = {}, acc = {};

    // Opaque byte offset ("always 0", but the compiler cannot prove it). Mutated by
    // empty asm every timestep so weight loads carry a loop-variant dependence:
    // LICM cannot hoist the (read-only) B fragments into 128+ persistent VGPRs
    // (which spilled to scratch in an earlier revision). Applying the opacity to an
    // offset instead of the pointer preserves pointer provenance, so these stay
    // global_load_b128 (LOADcnt only) rather than flat_load (LOADcnt+DScnt, which
    // contends with our real LDS spike traffic).
    size_t woff = 0;

    // ---- prologue: zero layer-3 partial acc (wave 0), async-stage X(0) (wave 15) --
    if (wave == 0) {
#pragma unroll
        for (int i = 0; i < 8; ++i) l3acc[lane * 8 + i] = 0.0f;
    }
    if (wave == NWAVE - 1) {
        const char* src0 = (const char*)(xT16 + ((size_t)0 * BB + b0) * CC);
#pragma unroll
        for (int j = 0; j < 8; ++j) {
            unsigned off  = (unsigned)((j * 32 + lane) * 16);
            unsigned ldsa = (unsigned)(LDS_XBUF + off);
            const char* ga = src0 + off;
            asm volatile("global_load_async_to_lds_b128 %0, %1, off"
                         :: "v"(ldsa), "v"(ga) : "memory");
        }
        asm volatile("s_wait_asynccnt 0" ::: "memory");
    }
    __syncthreads();

    for (int t = 0; t < TT; ++t) {
        // forbid hoisting weight loads across timesteps (see comment above)
        asm volatile("" : "+s"(woff));
        const _Float16* encp = enc16 + woff;
        const _Float16* hidp = hid16 + woff;
        const _Float16* owp  = ow16  + woff;

        const _Float16* xb = (const _Float16*)(lds_raw + LDS_XBUF + (t & 1) * 4096);

        // ---------------- layer 1: I1 = X_t @ enc_w^T  (f16 WMMA 16x16x32) ---------
        v8f c[4] = {};
        for (int kt = 0; kt < CC / 32; ++kt) {
            const int k0 = kt * 32;
            const _Float16* ap = xb + (size_t)nl * CC + k0 + half * 16;   // A row m = nl
            v8h alo = *(const v8h*)ap;
            v8h ahi = *(const v8h*)(ap + 8);
            v16h a = __builtin_shufflevector(alo, ahi, 0, 1, 2, 3, 4, 5, 6, 7,
                                             8, 9, 10, 11, 12, 13, 14, 15);
            const int kB = k0 + half * 16;
#pragma unroll
            for (int nt = 0; nt < 4; ++nt) {
                v16h b = *(const v16h*)(encp + (size_t)(h0 + nt * 16 + nl) * CC + kB);
                c[nt] = __builtin_amdgcn_wmma_f32_16x16x32_f16(
                    false, a, false, b, (short)0, c[nt], false, false);
            }
        }
#pragma unroll
        for (int nt = 0; nt < 4; ++nt) {
            const int h = h0 + nt * 16 + nl;
#pragma unroll
            for (int i = 0; i < 8; ++i) {
                float v = v1s[nt][i] + c[nt][i] + bias1[nt];
                float s = (v >= 1.0f) ? 1.0f : 0.0f;
                v1s[nt][i] = v - s * v;                 // hard reset to 0 on spike
                s1[(size_t)(mbase + i) * HH1 + h] = (_Float16)s;
            }
        }
        __syncthreads();                                // B1

        // consume layer-3 partial sums of step t-1 (ds_adds ordered by B1)
        if (wave == 0 && t > 0) {
#pragma unroll
            for (int i = 0; i < 8; ++i) {
                float v = v3[i] + l3acc[lane * 8 + i] + bias3;
                float s = (v >= 1.0f) ? 1.0f : 0.0f;
                v3[i] = v - s * v;
                acc[i] += s;
                l3acc[lane * 8 + i] = 0.0f;             // rearm before B2(t)
            }
        }
        // async-stage X(t+1) into the alternate buffer; overlapped with layer 2
        if (wave == NWAVE - 1 && t + 1 < TT) {
            const char* src = (const char*)(xT16 + ((size_t)(t + 1) * BB + b0) * CC);
            unsigned ldsbase = (unsigned)(LDS_XBUF + ((t + 1) & 1) * 4096);
#pragma unroll
            for (int j = 0; j < 8; ++j) {
                unsigned off  = (unsigned)((j * 32 + lane) * 16);
                unsigned ldsa = ldsbase + off;
                const char* ga = src + off;
                asm volatile("global_load_async_to_lds_b128 %0, %1, off"
                             :: "v"(ldsa), "v"(ga) : "memory");
            }
        }

        // ---------------- layer 2: I2 = S1 @ hid_w^T  (f16 WMMA 16x16x32) ----------
#pragma unroll
        for (int nt = 0; nt < 4; ++nt) c[nt] = (v8f){};
        for (int kt = 0; kt < HH1 / 32; ++kt) {
            const int k0 = kt * 32;
            const _Float16* ap = s1 + (size_t)nl * HH1 + k0 + half * 16;
            v8h alo = *(const v8h*)ap;
            v8h ahi = *(const v8h*)(ap + 8);
            v16h a = __builtin_shufflevector(alo, ahi, 0, 1, 2, 3, 4, 5, 6, 7,
                                             8, 9, 10, 11, 12, 13, 14, 15);
            const int kB = k0 + half * 16;
#pragma unroll
            for (int nt = 0; nt < 4; ++nt) {
                v16h b = *(const v16h*)(hidp + (size_t)(h0 + nt * 16 + nl) * HH1 + kB);
                c[nt] = __builtin_amdgcn_wmma_f32_16x16x32_f16(
                    false, a, false, b, (short)0, c[nt], false, false);
            }
        }
#pragma unroll
        for (int nt = 0; nt < 4; ++nt) {
            const int h = h0 + nt * 16 + nl;
#pragma unroll
            for (int i = 0; i < 8; ++i) {
                float v = v2s[nt][i] + c[nt][i] + bias2[nt];
                float s = (v >= 1.0f) ? 1.0f : 0.0f;
                v2s[nt][i] = v - s * v;
                s2[(size_t)(mbase + i) * HH2 + h] = (_Float16)s;
            }
        }
        if (wave == NWAVE - 1 && t + 1 < TT)            // X(t+1) must land before B2
            asm volatile("s_wait_asynccnt 0" ::: "memory");
        __syncthreads();                                // B2

        // ------- layer 3 partials: waves 0..3, 8 k-tiles each, ds_add_f32 ---------
        if (wave < 4) {
            v8f c3 = {};
            for (int kt = wave * 8; kt < wave * 8 + 8; ++kt) {
                const int k0 = kt * 32;
                const _Float16* ap = s2 + (size_t)nl * HH2 + k0 + half * 16;
                v8h alo = *(const v8h*)ap;
                v8h ahi = *(const v8h*)(ap + 8);
                v16h a = __builtin_shufflevector(alo, ahi, 0, 1, 2, 3, 4, 5, 6, 7,
                                                 8, 9, 10, 11, 12, 13, 14, 15);
                v16h b = *(const v16h*)(owp + (size_t)nl * HH2 + k0 + half * 16);
                c3 = __builtin_amdgcn_wmma_f32_16x16x32_f16(
                    false, a, false, b, (short)0, c3, false, false);
            }
#pragma unroll
            for (int i = 0; i < 8; ++i)
                atomicAdd(&l3acc[lane * 8 + i], c3[i]);
        }
    }

    // drain the last pending layer-3 step
    __syncthreads();
    if (wave == 0) {
#pragma unroll
        for (int i = 0; i < 8; ++i) {
            float v = v3[i] + l3acc[lane * 8 + i] + bias3;
            float s = (v >= 1.0f) ? 1.0f : 0.0f;
            v3[i] = v - s * v;
            acc[i] += s;
        }
        if (nl < NCC) {
#pragma unroll
            for (int i = 0; i < 8; ++i)
                out[(size_t)(b0 + mbase + i) * NCC + nl] = acc[i] * (1.0f / (float)TT);
        }
    }
}

// ------------- launcher -------------
extern "C" void kernel_launch(void* const* d_in, const int* in_sizes, int n_in,
                              void* d_out, int out_size, void* d_ws, size_t ws_size,
                              hipStream_t stream) {
    const float* x     = (const float*)d_in[0];   // [256,128,64]
    const float* enc_w = (const float*)d_in[1];   // [1024,128]
    const float* enc_b = (const float*)d_in[2];   // [1024]
    const float* hid_w = (const float*)d_in[3];   // [1024,1024]
    const float* hid_b = (const float*)d_in[4];   // [1024]
    const float* out_w = (const float*)d_in[5];   // [10,1024]
    const float* out_b = (const float*)d_in[6];   // [10]
    float* out = (float*)d_out;                   // [256,10]

    char* ws = (char*)d_ws;
    _Float16* xT16  = (_Float16*)ws;                       // 4 MB: [T,B,C] f16
    _Float16* hid16 = (_Float16*)(ws + (4u << 20));        // 2 MB
    _Float16* enc16 = (_Float16*)(ws + (6u << 20));        // 256 KB
    _Float16* ow16  = (_Float16*)(ws + (6u << 20) + (512u << 10)); // 32 KB

    snn_transpose_x<<<(BB * CC * TT + 255) / 256, 256, 0, stream>>>(x, xT16);
    snn_cvt16<<<(HH1 * HH2 + 255) / 256, 256, 0, stream>>>(hid_w, hid16, HH1 * HH2);
    snn_cvt16<<<(HH1 * CC + 255) / 256, 256, 0, stream>>>(enc_w, enc16, HH1 * CC);
    snn_build_ow<<<(16 * HH2 + 255) / 256, 256, 0, stream>>>(out_w, ow16);

    snn_fused<<<dim3(BB / BTILE), dim3(32 * NWAVE), LDS_BYTES, stream>>>(
        xT16, enc16, enc_b, hid16, hid_b, ow16, out_b, out);
}
